// LTE_12481174962985
// MI455X (gfx1250) — matrix operand order repaired and checked by reading, not compiled, not run on hardware
//
#include <hip/hip_runtime.h>
#include <math.h>

// ---------------- problem constants ----------------
#define Bb    4
#define CIN   64
#define Hh    128
#define Ww    128
#define HIDN  256
#define K9    (CIN * 9)        // 576 (GEMM K)
#define HW    (Hh * Ww)        // 16384
#define NTOT  (Bb * HW)        // 65536  (GEMM N)
#define MTOT  (2 * HIDN)       // 512    (coef rows 0..255, freq rows 256..511)

// GEMM tiling
#define MT    128
#define NT    128
#define KT    32
#define NKS   (K9 / KT)        // 18 K-steps
#define LDSS  40               // padded LDS row stride (bf16 elems): 128x40 tile = 5120 elems = 10240 B
#define TILE_E (MT * LDSS)     // 5120 elements per packed A tile
#define TILE_B (TILE_E * 2)    // 10240 bytes

// padded input image: [B][C][130][132], interior at (+1,+1)
#define YP    130
#define XP    132
#define PLANE (YP * XP)        // 17160 elements

typedef __attribute__((ext_vector_type(16))) __bf16 v16bf;
typedef __attribute__((ext_vector_type(8)))  float  v8f;
typedef __attribute__((ext_vector_type(4)))  float  f32x4;
typedef __attribute__((ext_vector_type(4)))  unsigned int u32x4;
typedef __attribute__((ext_vector_type(8)))  int    i32x8;
typedef __attribute__((ext_vector_type(4)))  int    i32x4;

// f32 -> bf16, round-to-nearest-even
__device__ __forceinline__ unsigned short f2bf(float f) {
  union { float f; unsigned u; } v; v.f = f;
  unsigned u = v.u;
  u += 0x7FFFu + ((u >> 16) & 1u);
  return (unsigned short)(u >> 16);
}

// ---------------- weight pack kernel ----------------
// Apack[mb][ks][row][col(40)] : col<32 -> K = ks*32+col with tap-major K order
//   k = tap*64 + c  (tap = 3*dy+dx, c = input channel), rows padded with zeros.
__global__ void pack_weights(const float* __restrict__ coef_w,
                             const float* __restrict__ freq_w,
                             unsigned short* __restrict__ Apack) {
  int i = blockIdx.x * 256 + threadIdx.x;      // [0, 4*18*5120)
  if (i >= (MTOT / MT) * NKS * TILE_E) return;
  int mb = i / (NKS * TILE_E);
  int r  = i - mb * (NKS * TILE_E);
  int ks = r / TILE_E;
  int e  = r - ks * TILE_E;
  int row = e / LDSS;
  int col = e - row * LDSS;
  float v = 0.0f;
  if (col < KT) {
    int k   = ks * KT + col;       // tap-major K
    int tap = k >> 6;              // 0..8
    int c   = k & 63;
    int m   = mb * MT + row;       // 0..511
    int src = c * 9 + tap;         // source inner index within [C][3][3]
    v = (m < HIDN) ? coef_w[m * K9 + src] : freq_w[(m - HIDN) * K9 + src];
  }
  Apack[i] = f2bf(v);
}

// ---------------- input pad+convert kernel ----------------
__global__ void pad_input(const float* __restrict__ inp,
                          unsigned short* __restrict__ Ipad, int n) {
  int i = blockIdx.x * 256 + threadIdx.x;
  if (i >= n) return;
  int x  = i & (Ww - 1);
  int y  = (i >> 7) & (Hh - 1);
  int cc = i >> 14;                             // b*CIN + c
  Ipad[(size_t)cc * PLANE + (size_t)(y + 1) * XP + (x + 1)] = f2bf(inp[i]);
}

// ---------------- fused dual-conv as WMMA GEMM ----------------
// C[m][n] = sum_k A[m][k]*B[k][n] + bias[m];  C = f32 [512][65536] (M-major)
// LDS (dynamic, 40960 B): As0 | As1 | Bs0 | Bs1, each 5120 ushort.
__launch_bounds__(256)
__global__ void conv_gemm(const unsigned short* __restrict__ Apack,
                          const unsigned short* __restrict__ Ipad,
                          float* __restrict__ Cout,
                          const float* __restrict__ coef_b,
                          const float* __restrict__ freq_b) {
  extern __shared__ unsigned short smem[];     // no static LDS -> dynamic base offset 0

  const int tid  = threadIdx.x;
  const int lane = tid & 31;
  const int wave = tid >> 5;        // 8 waves
  const int wm   = wave & 3;        // m-frags {2wm, 2wm+1}
  const int wn   = wave >> 2;       // n-frags {4wn .. 4wn+3}

  const int n0 = blockIdx.x * NT;   // one full image row per N-tile (W=128)
  const int mb = blockIdx.y;        // m-block 0..3
  const int m0 = mb * MT;
  const int b  = n0 >> 14;
  const int y  = (n0 >> 7) & 127;

  v8f acc[2][4];
#pragma unroll
  for (int i = 0; i < 2; ++i)
#pragma unroll
    for (int j = 0; j < 4; ++j) acc[i][j] = (v8f){};

  // B staging mapping: thread -> (x, k-half)
  const int bx  = tid & 127;
  const int bkh = tid >> 7;                    // 0/1 -> k_local in [16*bkh, 16*bkh+16)

  // stage K-step `ks` into buffer `buf`
  auto stage = [&](int ks, int buf) {
    // ---- B tile: on-the-fly im2col from padded input (no bounds checks) ----
    const int k0    = ks * KT;
    const int tap   = k0 >> 6;                 // uniform per K-step
    const int cbase = (k0 & 63) + bkh * 16;    // first channel of this thread's 16
    const int dy    = tap / 3;                 // scalar
    const int dx    = tap - dy * 3;            // scalar
    const unsigned short* src =
        Ipad + (size_t)(b * CIN + cbase) * PLANE + (size_t)(y + dy) * XP + (bx + dx);
    union { unsigned short s[16]; f32x4 q[2]; } u;
#pragma unroll
    for (int i = 0; i < 16; ++i) u.s[i] = src[(size_t)i * PLANE];
    unsigned short* BsBuf = smem + 2 * TILE_E + buf * TILE_E;
    *(f32x4*)(&BsBuf[bx * LDSS + bkh * 16])     = u.q[0];
    *(f32x4*)(&BsBuf[bx * LDSS + bkh * 16 + 8]) = u.q[1];

    // ---- A tile: contiguous packed 10240 B via Tensor Data Mover ----
#if __has_builtin(__builtin_amdgcn_tensor_load_to_lds)
    if (wave == 0) {
      // TDM: 1-D tile, data_size=2B, tile_dim0=5120 elems
      unsigned long long ga =
          (unsigned long long)(const void*)(Apack + ((size_t)mb * NKS + ks) * TILE_E);
      unsigned lds_addr = (unsigned)(buf * TILE_B);
      u32x4 g0 = { 1u,                                   // count=1
                   lds_addr,                             // [63:32] lds_addr
                   (unsigned)(ga & 0xFFFFFFFFu),         // [95:64] global_addr lo
                   (unsigned)((ga >> 32) & 0x1FFFFFFu) | 0x80000000u }; // addr hi | type=2
      i32x8 g1 = { (int)0x00010000,                      // data_size=1 (2 bytes)
                   (int)((TILE_E & 0xFFFF) << 16),       // tensor_dim0 lo16 @ [63:48]
                   (int)(1 << 16),                       // tensor_dim0 hi16=0 | tensor_dim1=1
                   (int)((TILE_E & 0xFFFF) << 16),       // tensor_dim1 hi=0 | tile_dim0 @ [127:112]
                   (int)1,                               // tile_dim1=1, tile_dim2=0
                   (int)TILE_E,                          // tensor_dim0_stride lo32
                   0, 0 };
      i32x4 g2 = { 0, 0, 0, 0 };
      i32x4 g3 = { 0, 0, 0, 0 };
      i32x8 g4 = { 0, 0, 0, 0, 0, 0, 0, 0 };             // unused groups (2D tensor)
      __builtin_amdgcn_tensor_load_to_lds(g0, g1, g2, g3, g4, 0);
    }
#else
    {
      const f32x4* srcA =
          (const f32x4*)(Apack + ((size_t)mb * NKS + ks) * TILE_E);
      f32x4* dstA = (f32x4*)(smem + buf * TILE_E);
#pragma unroll
      for (int idx = tid; idx < TILE_B / 16; idx += 256) dstA[idx] = srcA[idx];
    }
#endif
  };

  stage(0, 0);
#if __has_builtin(__builtin_amdgcn_tensor_load_to_lds)
  if (wave == 0) __builtin_amdgcn_s_wait_tensorcnt(0);
#endif
  __syncthreads();

  for (int ks = 0; ks < NKS; ++ks) {
    const int cur = ks & 1;
    if (ks + 1 < NKS) stage(ks + 1, cur ^ 1);

    const unsigned short* As = smem + cur * TILE_E;
    const unsigned short* Bs = smem + 2 * TILE_E + cur * TILE_E;

    // ---- fragments per ISA VGPR layouts ----
    const int arf = lane & 15;
    const int kA  = (lane >> 4) * 8;
    v16bf afrag[2];
#pragma unroll
    for (int i = 0; i < 2; ++i) {
      int mrow = (wm * 2 + i) * 16 + arf;
      union { v16bf v; f32x4 q[2]; } u;
      u.q[0] = *(const f32x4*)(&As[mrow * LDSS + kA]);
      u.q[1] = *(const f32x4*)(&As[mrow * LDSS + kA + 16]);
      afrag[i] = u.v;
    }
    const int ncf = lane & 15;
    const int kB  = (lane >> 4) * 8;
    v16bf bfrag[4];
#pragma unroll
    for (int j = 0; j < 4; ++j) {
      int nrow = (wn * 4 + j) * 16 + ncf;
      union { v16bf v; f32x4 q[2]; } u;
      u.q[0] = *(const f32x4*)(&Bs[nrow * LDSS + kB]);
      u.q[1] = *(const f32x4*)(&Bs[nrow * LDSS + kB + 16]);
      bfrag[j] = u.v;
    }

#pragma unroll
    for (int i = 0; i < 2; ++i)
#pragma unroll
      for (int j = 0; j < 4; ++j)
        acc[i][j] = __builtin_amdgcn_wmma_f32_16x16x32_bf16(
            false, afrag[i], false, bfrag[j], (short)0, acc[i][j], false, false);

#if __has_builtin(__builtin_amdgcn_tensor_load_to_lds)
    if (wave == 0) __builtin_amdgcn_s_wait_tensorcnt(0);
#endif
    __syncthreads();
  }

  // ---- epilogue: + bias, store f32 M-major ----
  const int mOff = (lane >> 4) * 8;
  const int nOff = lane & 15;
#pragma unroll
  for (int i = 0; i < 2; ++i)
#pragma unroll
    for (int j = 0; j < 4; ++j) {
      int mBase = m0 + (wm * 2 + i) * 16 + mOff;
      int nIdx  = n0 + (wn * 4 + j) * 16 + nOff;
#pragma unroll
      for (int v = 0; v < 8; ++v) {
        int m = mBase + v;
        float bias = (m < HIDN) ? coef_b[m] : freq_b[m - HIDN];
        Cout[(size_t)m * NTOT + nIdx] = acc[i][j][v] + bias;
      }
    }
}

// ---------------- LTE render epilogue ----------------
__launch_bounds__(256)
__global__ void lte_render(const float* __restrict__ conv,
                           const float* __restrict__ coord,
                           const float* __restrict__ cell,
                           const float* __restrict__ phase_w,
                           float* __restrict__ out) {
  const int tx = threadIdx.x;       // 0..31
  const int ty = threadIdx.y;       // 0..7
  int blk = blockIdx.x;
  const int xc = blk & 3;  blk >>= 2;
  const int y  = blk & 127; blk >>= 7;
  const int b  = blk;               // 0..3
  const int x  = xc * 32 + tx;
  const int q  = y * Ww + x;

  const float c0 = coord[((size_t)(b * HW + q)) * 2 + 0];
  const float c1 = coord[((size_t)(b * HW + q)) * 2 + 1];
  const float relc0 = cell[((size_t)(b * HW + q)) * 2 + 0] * (float)Hh;
  const float relc1 = cell[((size_t)(b * HW + q)) * 2 + 1] * (float)Ww;

  const float rx = 1.0f / (float)Hh;
  const float ry = 1.0f / (float)Ww;
  const float eps = 1e-6f;

  float rel0[4], rel1[4], areas[4];
  int   idxp[4];
  int cI = 0;
#pragma unroll
  for (int vxi = 0; vxi < 2; ++vxi)
#pragma unroll
    for (int vyi = 0; vyi < 2; ++vyi) {
      const float vx = vxi ? 1.0f : -1.0f;
      const float vy = vyi ? 1.0f : -1.0f;
      float cc0 = c0 + vx * rx + eps;
      float cc1 = c1 + vy * ry + eps;
      cc0 = fminf(fmaxf(cc0, -1.0f + 1e-6f), 1.0f - 1e-6f);
      cc1 = fminf(fmaxf(cc1, -1.0f + 1e-6f), 1.0f - 1e-6f);
      float iyf = rintf(((cc0 + 1.0f) * (float)Hh - 1.0f) * 0.5f);
      float ixf = rintf(((cc1 + 1.0f) * (float)Ww - 1.0f) * 0.5f);
      int iy = (int)fminf(fmaxf(iyf, 0.0f), (float)(Hh - 1));
      int ix = (int)fminf(fmaxf(ixf, 0.0f), (float)(Ww - 1));
      idxp[cI] = iy * Ww + ix;
      float qc0 = -1.0f + (2.0f * (float)iy + 1.0f) / (float)Hh;
      float qc1 = -1.0f + (2.0f * (float)ix + 1.0f) / (float)Ww;
      rel0[cI] = (c0 - qc0) * (float)Hh;
      rel1[cI] = (c1 - qc1) * (float)Ww;
      areas[cI] = fabsf(rel0[cI] * rel1[cI]);
      ++cI;
    }
  const float tot    = areas[0] + areas[1] + areas[2] + areas[3] + 4e-9f;
  const float weight = areas[3] / tot;   // faithful: last corner's raw area only

  const float* coefB = conv;                          // m in [0,256)
  const float* freqB = conv + (size_t)HIDN * NTOT;    // m in [256,512)
  const size_t colBase = (size_t)b * HW;
  const float  PIv = 3.14159265358979323846f;

#pragma unroll 4
  for (int kk = 0; kk < 16; ++kk) {
    const int j = ty + 8 * kk;                        // 0..127 frequency pair
    const float ph = relc0 * phase_w[2 * j] + relc1 * phase_w[2 * j + 1];
    float accC = 0.0f, accS = 0.0f;
#pragma unroll
    for (int ci = 0; ci < 4; ++ci) {
      const size_t col = colBase + (size_t)idxp[ci];
      const float f0 = freqB[(size_t)(2 * j)     * NTOT + col];
      const float f1 = freqB[(size_t)(2 * j + 1) * NTOT + col];
      const float qc = coefB[(size_t)j           * NTOT + col];
      const float qs = coefB[(size_t)(j + 128)   * NTOT + col];
      const float fv = f0 * rel0[ci] + f1 * rel1[ci] + ph;
      float s, c;
      __sincosf(PIv * fv, &s, &c);
      accC += qc * c;
      accS += qs * s;
    }
    out[((size_t)(b * HIDN + j))       * HW + q] = accC * weight;
    out[((size_t)(b * HIDN + 128 + j)) * HW + q] = accS * weight;
  }
}

// ---------------- launch ----------------
extern "C" void kernel_launch(void* const* d_in, const int* in_sizes, int n_in,
                              void* d_out, int out_size, void* d_ws, size_t ws_size,
                              hipStream_t stream) {
  const float* inp     = (const float*)d_in[0];
  const float* coord   = (const float*)d_in[1];
  const float* cell    = (const float*)d_in[2];
  const float* coef_w  = (const float*)d_in[3];
  const float* coef_b  = (const float*)d_in[4];
  const float* freq_w  = (const float*)d_in[5];
  const float* freq_b  = (const float*)d_in[6];
  const float* phase_w = (const float*)d_in[7];
  float* out = (float*)d_out;

  // workspace layout (bytes):
  //   [0,        737280)   : packed bf16 weights, 4 x 18 x (128x40) tiles
  //   [737280,  9523200)   : padded bf16 input [4][64][130][132]
  //   [9523200, 143740928) : f32 conv out [512][65536]
  unsigned short* Apack = (unsigned short*)d_ws;
  unsigned short* Ipad  = (unsigned short*)((char*)d_ws + 737280);
  float*          convO = (float*)((char*)d_ws + 9523200);

  const int nApack = (MTOT / MT) * NKS * TILE_E;        // 368640
  const int nIn    = Bb * CIN * Hh * Ww;                // 4194304

  (void)hipMemsetAsync(Ipad, 0, (size_t)Bb * CIN * PLANE * 2, stream);  // zero border
  pack_weights<<<(nApack + 255) / 256, 256, 0, stream>>>(coef_w, freq_w, Apack);
  pad_input<<<(nIn + 255) / 256, 256, 0, stream>>>(inp, Ipad, nIn);
  conv_gemm<<<dim3(NTOT / NT, MTOT / MT), 256, 4 * TILE_B, stream>>>(
      Apack, Ipad, convO, coef_b, freq_b);
  lte_render<<<Bb * Hh * (Ww / 32), dim3(32, 8), 0, stream>>>(convO, coord, cell, phase_w, out);
}
